// CTCLSTM_15504831938822
// MI455X (gfx1250) — matrix operand order repaired and checked by reading, not compiled, not run on hardware
//
#include <hip/hip_runtime.h>
#include <hip/hip_bf16.h>
#include <math.h>

#define T_DIM 2048
#define B_DIM 256
#define V_DIM 64
#define H_DIM 256
#define BLANK 63
#define G4H   1024
#define HPAD  264            // 256 + 8 bf16 pad -> 528B row stride, spreads LDS banks
#define ROWS  32             // batch rows per workgroup
#define NWG   (B_DIM / ROWS) // 8 workgroups
// dynamic LDS: h double buffer + 4 K-slices of W_hh (bf16 fragment tiles)
#define HBUF_ELEMS (2 * ROWS * HPAD)                   // 16896 bf16 = 33792 B
#define WLDS_ELEMS (4 * 64 * 32 * 16)                  // 131072 bf16 = 256 KB
#define SHBYTES    ((HBUF_ELEMS + WLDS_ELEMS) * 2)     // 295936 B < 320 KB

typedef __attribute__((ext_vector_type(16))) __bf16 bf16x16;
typedef __attribute__((ext_vector_type(8)))  __bf16 bf16x8;
typedef __attribute__((ext_vector_type(8)))  float  f32x8;

__device__ __forceinline__ float sigmoidf_(float x) {
    return 1.0f / (1.0f + __expf(-x));
}
__device__ __forceinline__ float tanhf_(float x) {
    // tanh(x) = 2*sigmoid(2x) - 1 : one v_exp_f32 + v_rcp_f32
    return 2.0f / (1.0f + __expf(-2.0f * x)) - 1.0f;
}

// ---------------------------------------------------------------------------
// Kernel A: per-(t,b) argmax over V=64 logits (softmax skipped: monotonic).
// ---------------------------------------------------------------------------
__global__ void argmax_kernel(const float* __restrict__ x, int* __restrict__ preds) {
    int gid = blockIdx.x * blockDim.x + threadIdx.x;  // t*B + b
    const float4* row = (const float4*)(x + (size_t)gid * V_DIM);
    float best = -3.402823466e+38f;
    int bi = 0;
#pragma unroll
    for (int i = 0; i < 16; ++i) {
        float4 v = row[i];
        if (v.x > best) { best = v.x; bi = 4 * i + 0; }
        if (v.y > best) { best = v.y; bi = 4 * i + 1; }
        if (v.z > best) { best = v.z; bi = 4 * i + 2; }
        if (v.w > best) { best = v.w; bi = 4 * i + 3; }
    }
    preds[gid] = bi;
}

// ---------------------------------------------------------------------------
// Kernel B: CTC unique-consecutive/de-blank compaction. One thread per batch
// column; preds[t*B+b] loads are fully coalesced. Tail-fills BLANK.
// ---------------------------------------------------------------------------
__global__ void ctc_compact_kernel(const int* __restrict__ preds, int* __restrict__ labels) {
    int b = threadIdx.x;  // 256 threads, 1 block
    int prev = -1, cnt = 0;
    for (int t = 0; t < T_DIM; ++t) {
        int p = preds[t * B_DIM + b];
        if (p != BLANK && (t == 0 || p != prev)) {
            labels[cnt * B_DIM + b] = p;
            ++cnt;
        }
        prev = p;
    }
    for (int t = cnt; t < T_DIM; ++t) labels[t * B_DIM + b] = BLANK;
}

// ---------------------------------------------------------------------------
// Kernel C: gate LUT  G[v][j] = dot(emb[v,:], W_ih[j,:]) + b_ih[j] + b_hh[j]
// (64 x 1024 f32 = 256KB, L2 resident). Deletes the T*B input-proj GEMM.
// ---------------------------------------------------------------------------
__global__ void build_g_kernel(const float* __restrict__ emb,
                               const float* __restrict__ W_ih,
                               const float* __restrict__ b_ih,
                               const float* __restrict__ b_hh,
                               float* __restrict__ G) {
    __shared__ float e[H_DIM];
    int v = blockIdx.x;
    e[threadIdx.x] = emb[(size_t)v * H_DIM + threadIdx.x];
    __syncthreads();
    for (int j = threadIdx.x; j < G4H; j += 256) {
        const float* w = W_ih + (size_t)j * H_DIM;
        float acc = 0.f;
#pragma unroll 4
        for (int k = 0; k < H_DIM; ++k) acc = fmaf(e[k], w[k], acc);
        G[(size_t)v * G4H + j] = acc + b_ih[j] + b_hh[j];
    }
}

// ---------------------------------------------------------------------------
// Kernel P: pack W (rows x 256 f32) into bf16 WMMA B-fragment tiles.
// Tile (nt,kt) = 32x16 slice B[k][n] = W[nt*16+n][kt*32+k]; per-lane layout:
// lane l holds column n=l%16, K = 16*(l/16)..+15 -> 32 contiguous bytes/lane.
// ---------------------------------------------------------------------------
__global__ void pack_wtiles_kernel(const float* __restrict__ W,
                                   __bf16* __restrict__ out, int ntiles) {
    int tid = blockIdx.x * blockDim.x + threadIdx.x;
    if (tid >= ntiles * 8 * 32) return;
    int lane = tid & 31;
    int kt   = (tid >> 5) & 7;
    int nt   = tid >> 8;
    int n  = lane & 15;
    int kb = (lane >> 4) * 16;
    const float* src = W + (size_t)(nt * 16 + n) * H_DIM + kt * 32 + kb;
    __bf16* dst = out + (size_t)tid * 16;
#pragma unroll
    for (int j = 0; j < 16; ++j) dst[j] = (__bf16)src[j];
}

// ---------------------------------------------------------------------------
// Kernel D: persistent LSTM + fused output projection.
// 8 blocks x 512 threads (16 wave32), one 32-row batch tile per block; no
// inter-workgroup sync across the 2048-step recurrence (B is the parallel
// axis). Wave w owns h/c columns [16w,16w+16): computes gate N-tiles
// {w,16+w,32+w,48+w} for both 16-row M-tiles (each B fragment feeds 2 WMMAs);
// nonlinearity is wave-local; c lives in registers in C-fragment layout; h_t
// is double-buffered in LDS. W_hh K-tiles 0..3 (256 KB bf16) are LDS-resident
// (ds_load_b128); K-tiles 4..7 stream from L2 (global_load_b128).
// One opaque 32-bit base index per K-tile (asm "+v") blocks LICM (no spill)
// while the 4 gate fragments ride on load IMMEDIATE offsets (16KB LDS steps /
// 128KB imm24 global steps), minimizing per-fragment address VALU and the
// WMMA->VALU WAR hazard NOPs seen in round 3.
// Waves 0..7 also emit out[t] = h_t @ W_out^T + b_out.
// ---------------------------------------------------------------------------
__global__ void __launch_bounds__(512) lstm_kernel(
    const int*    __restrict__ labels,
    const float*  __restrict__ G,
    const __bf16* __restrict__ WhhT,   // 64 nt x 8 kt packed tiles
    const __bf16* __restrict__ WoutT,  // 4 nt x 8 kt packed tiles
    const float*  __restrict__ b_out,
    float*        __restrict__ out) {
    extern __shared__ __align__(16) char smem[];
    __bf16* sbase = (__bf16*)smem;
    __bf16* hbuf  = sbase;                       // [2][ROWS][HPAD]
    __bf16* wlds  = sbase + HBUF_ELEMS;          // [4][64][32*16]

    const int tid  = threadIdx.x;
    const int w    = tid >> 5;   // wave 0..15
    const int lane = tid & 31;
    const int n    = lane & 15;  // fragment column / A row-within-tile
    const int half = lane >> 4;
    const int b0   = blockIdx.x * ROWS;
    const int kcol = w * 16 + n; // this wave's h/c column for lane

    // h_{-1} = 0
    for (int i = tid; i < HBUF_ELEMS; i += 512) hbuf[i] = (__bf16)0.0f;
    // stage W_hh K-tiles 0..3 into LDS (one-time 256 KB copy, 16B vectors)
    for (int g = tid; g < 4 * 64 * 32; g += 512) {
        int sl = g & 31, snt = (g >> 5) & 63, skt = g >> 11;
        const uint4* src = (const uint4*)(WhhT + (size_t)((snt * 8 + skt) * 32 + sl) * 16);
        uint4* dst = (uint4*)(wlds + (size_t)g * 16);
        dst[0] = src[0];
        dst[1] = src[1];
    }
    __syncthreads();

    f32x8 c[2] = {{}, {}};  // cell state, elem r of M-tile mt -> row mt*16 + half*8 + r
    const int onx = w & 3, omt = w >> 2;
    float bout = (w < 8) ? b_out[onx * 16 + n] : 0.0f;

    int buf = 0;
    for (int t = 0; t < T_DIM; ++t) {
        __bf16* hrd = hbuf + buf * ROWS * HPAD;
        // ---- gates = h_{t-1} @ W_hh^T : K=256 in 8 tiles, 4 gates, 2 M-tiles
        f32x8 acc[4][2] = {{{}, {}}, {{}, {}}, {{}, {}}, {{}, {}}};
#pragma unroll
        for (int kt = 0; kt < 8; ++kt) {
            bf16x16 a[2];
#pragma unroll
            for (int mt = 0; mt < 2; ++mt) {
                const __bf16* hrow = hrd + (mt * 16 + n) * HPAD + kt * 32 + half * 16;
                bf16x8 lo = *(const bf16x8*)hrow;
                bf16x8 hi = *(const bf16x8*)(hrow + 8);
#pragma unroll
                for (int i = 0; i < 8; ++i) { a[mt][i] = lo[i]; a[mt][i + 8] = hi[i]; }
            }
            bf16x16 bfr[4];
            if (kt < 4) {
                // absolute element index into smem; gate stride = 8192 elem (16KB)
                int sidx = HBUF_ELEMS + kt * (64 * 512) + (w * 32 + lane) * 16;
                asm("" : "+v"(sidx));  // opaque: stays in-loop, stays LDS
                const __bf16* wp = sbase + sidx;
#pragma unroll
                for (int g = 0; g < 4; ++g)
                    bfr[g] = *(const bf16x16*)(wp + g * 8192);
            } else {
                // gate stride = 16 ntiles * 8 kt * 512 elem = 65536 elem (128KB imm24)
                int gidx = ((w * 8 + kt) * 32 + lane) * 16;
                asm("" : "+v"(gidx));  // opaque: stays in-loop, stays global
                const __bf16* gp = WhhT + gidx;
#pragma unroll
                for (int g = 0; g < 4; ++g)
                    bfr[g] = *(const bf16x16*)(gp + g * 65536);
            }
#pragma unroll
            for (int g = 0; g < 4; ++g) {
                acc[g][0] = __builtin_amdgcn_wmma_f32_16x16x32_bf16(
                    false, a[0], false, bfr[g], (short)0, acc[g][0], false, false);
                acc[g][1] = __builtin_amdgcn_wmma_f32_16x16x32_bf16(
                    false, a[1], false, bfr[g], (short)0, acc[g][1], false, false);
            }
        }

        // ---- add LUT gates, nonlinearity, update c, write h_t to LDS
        const int wb = buf ^ 1;
        __bf16* hwr = hbuf + wb * ROWS * HPAD;
#pragma unroll
        for (int mt = 0; mt < 2; ++mt) {
#pragma unroll
            for (int r = 0; r < 8; ++r) {
                int m = mt * 16 + half * 8 + r;
                int lbl = labels[t * B_DIM + b0 + m];
                const float* gr = G + (size_t)lbl * G4H + kcol;
                float gi = acc[0][mt][r] + gr[0];
                float gf = acc[1][mt][r] + gr[256];
                float gg = acc[2][mt][r] + gr[512];
                float go = acc[3][mt][r] + gr[768];
                float cn = sigmoidf_(gf) * c[mt][r] + sigmoidf_(gi) * tanhf_(gg);
                c[mt][r] = cn;
                float hv = sigmoidf_(go) * tanhf_(cn);
                hwr[m * HPAD + kcol] = (__bf16)hv;
            }
        }
        __syncthreads();  // h_t visible; all readers of hbuf[buf] are done

        // ---- fused out[t] = h_t @ W_out^T + b_out : waves 0..7, (nt,Mtile)
        if (w < 8) {
            f32x8 oacc;
#pragma unroll
            for (int r = 0; r < 8; ++r) oacc[r] = bout;
#pragma unroll
            for (int kt = 0; kt < 8; ++kt) {
                const __bf16* hrow = hwr + (omt * 16 + n) * HPAD + kt * 32 + half * 16;
                bf16x8 lo = *(const bf16x8*)hrow;
                bf16x8 hi = *(const bf16x8*)(hrow + 8);
                bf16x16 a;
#pragma unroll
                for (int i = 0; i < 8; ++i) { a[i] = lo[i]; a[i + 8] = hi[i]; }
                int oidx = ((onx * 8 + kt) * 32 + lane) * 16;
                asm("" : "+v"(oidx));
                const bf16x16 bfr = *(const bf16x16*)(WoutT + oidx);
                oacc = __builtin_amdgcn_wmma_f32_16x16x32_bf16(
                    false, a, false, bfr, (short)0, oacc, false, false);
            }
            float* orow = out + ((size_t)t * B_DIM + b0 + omt * 16) * V_DIM + onx * 16 + n;
#pragma unroll
            for (int r = 0; r < 8; ++r)
                orow[(size_t)(half * 8 + r) * V_DIM] = oacc[r];
        }
        buf = wb;
    }
}

// ---------------------------------------------------------------------------
extern "C" void kernel_launch(void* const* d_in, const int* in_sizes, int n_in,
                              void* d_out, int out_size, void* d_ws, size_t ws_size,
                              hipStream_t stream) {
    (void)in_sizes; (void)n_in; (void)out_size; (void)ws_size;
    const float* x     = (const float*)d_in[0];
    const float* emb   = (const float*)d_in[1];
    const float* W_ih  = (const float*)d_in[2];
    const float* W_hh  = (const float*)d_in[3];
    const float* b_ih  = (const float*)d_in[4];
    const float* b_hh  = (const float*)d_in[5];
    const float* W_out = (const float*)d_in[6];
    const float* b_out = (const float*)d_in[7];
    float* out = (float*)d_out;

    char* ws = (char*)d_ws;
    const size_t tb_bytes = (size_t)T_DIM * B_DIM * sizeof(int);  // 2 MB
    int*    preds  = (int*)(ws);
    int*    labels = (int*)(ws + tb_bytes);
    float*  G      = (float*)(ws + 2 * tb_bytes);                       // 256 KB
    __bf16* WhhT   = (__bf16*)(ws + 2 * tb_bytes + (size_t)V_DIM * G4H * 4);
    __bf16* WoutT  = (__bf16*)((char*)WhhT + (size_t)G4H * H_DIM * 2);  // +512 KB

    (void)hipFuncSetAttribute((const void*)lstm_kernel,
                              hipFuncAttributeMaxDynamicSharedMemorySize, SHBYTES);

    argmax_kernel<<<(T_DIM * B_DIM) / 256, 256, 0, stream>>>(x, preds);
    ctc_compact_kernel<<<1, B_DIM, 0, stream>>>(preds, labels);
    build_g_kernel<<<V_DIM, 256, 0, stream>>>(emb, W_ih, b_ih, b_hh, G);
    pack_wtiles_kernel<<<64, 256, 0, stream>>>(W_hh, WhhT, 64);
    pack_wtiles_kernel<<<4, 256, 0, stream>>>(W_out, WoutT, 4);
    lstm_kernel<<<NWG, 512, SHBYTES, stream>>>(labels, G, WhhT, WoutT, b_out, out);
}
